// ROIClassifierHead_18330920419785
// MI455X (gfx1250) — matrix operand order repaired and checked by reading, not compiled, not run on hardware
//
#include <hip/hip_runtime.h>

// ---------------------------------------------------------------------------
// ROI classifier head = 3-layer MLP on 2048 rows (identity ROI permutation):
//   H1 = relu(X  @ W1 + b1)   X:[2048,12544]  W1:[12544,1024]
//   H2 = relu(H1 @ W2 + b2)   W2:[1024,1024]
//   Y  =      H2 @ Wc + bc    Wc:[1024,81]
//
// MI455X / gfx1250 (wave32):
//  - pre-convert X -> bf16 [M][K]; W* -> bf16 TRANSPOSED [N][K] (one pass)
//  - pure-bf16 GEMM: v_wmma_f32_16x16x32_bf16, fp32 accumulation
//  - 128x128 block tile, 8 waves (2x4), 64x32 per wave, TILE_K=64
//  - TRIPLE-buffered LDS filled by GLOBAL_LOAD_ASYNC_TO_LDS_B128 (no staging
//    VGPRs -> no spills), synchronized with s_wait_asynccnt + one barrier/tile
//  - H1/H2 kept in bf16; final logits stored fp32
// ---------------------------------------------------------------------------

typedef __bf16 bf16_t;
typedef bf16_t v16bf __attribute__((ext_vector_type(16)));
typedef float  v8f   __attribute__((ext_vector_type(8)));

#define TILE_M 128
#define TILE_N 128
#define TILE_K 64
#define LDSK   72   // 64 + 8 pad bf16 elems; 144 B row stride, 16B-aligned

__device__ __forceinline__ unsigned int pack2_bf16(float a, float b) {
  unsigned int ua = __float_as_uint(a);
  unsigned int ub = __float_as_uint(b);
  ua = (ua + 0x7FFFu + ((ua >> 16) & 1u)) >> 16;   // RNE
  ub = (ub + 0x7FFFu + ((ub >> 16) & 1u)) >> 16;
  return (ua & 0xFFFFu) | (ub << 16);
}
__device__ __forceinline__ unsigned short f2bf(float a) {
  unsigned int ua = __float_as_uint(a);
  return (unsigned short)((ua + 0x7FFFu + ((ua >> 16) & 1u)) >> 16);
}

union FragU { uint4 u[2]; v16bf v; };

// one 16-byte async global->LDS copy per lane (ASYNCcnt++)
__device__ __forceinline__ void async_copy_b128(const void* gsrc, void* ldst) {
  unsigned lds_addr = (unsigned)(unsigned long long)ldst;  // flat->LDS: low 32 bits
  asm volatile("global_load_async_to_lds_b128 %0, %1, off"
               :: "v"(lds_addr), "v"(gsrc) : "memory");
}

// ---------------- elementwise fp32 -> bf16 (for X) -------------------------
__global__ __launch_bounds__(256) void cvt_bf16_kernel(
    const float* __restrict__ in, unsigned short* __restrict__ out, long n4) {
  long i = (long)blockIdx.x * 256 + threadIdx.x;
  if (i < n4) {
    float4 v = ((const float4*)in)[i];
    ((uint2*)out)[i] = make_uint2(pack2_bf16(v.x, v.y), pack2_bf16(v.z, v.w));
  }
}

// ---------------- W[K][N] fp32 -> Wt[N][K] bf16, 64x64 tiles ---------------
__global__ __launch_bounds__(256) void cvt_transpose_kernel(
    const float* __restrict__ W, unsigned short* __restrict__ Wt, int K, int N) {
  __shared__ unsigned short t[64 * LDSK];          // t[n][k]
  const int tid = threadIdx.x;
  const int k0 = blockIdx.y * 64;
  const int n0 = blockIdx.x * 64;
  const bool nFull = (n0 + 64) <= N;

#pragma unroll
  for (int i = 0; i < 4; ++i) {
    int f  = tid + i * 256;        // 1024 float4 = 64 k-rows x 16 float4
    int kr = f >> 4;
    int c4 = f & 15;
    float4 v;
    if (nFull) {
      v = *(const float4*)(W + (size_t)(k0 + kr) * N + n0 + c4 * 4);
    } else {
      int nb = n0 + c4 * 4;
      v.x = (nb + 0 < N) ? W[(size_t)(k0 + kr) * N + nb + 0] : 0.0f;
      v.y = (nb + 1 < N) ? W[(size_t)(k0 + kr) * N + nb + 1] : 0.0f;
      v.z = (nb + 2 < N) ? W[(size_t)(k0 + kr) * N + nb + 2] : 0.0f;
      v.w = (nb + 3 < N) ? W[(size_t)(k0 + kr) * N + nb + 3] : 0.0f;
    }
    int cb = c4 * 4;
    t[(cb + 0) * LDSK + kr] = f2bf(v.x);
    t[(cb + 1) * LDSK + kr] = f2bf(v.y);
    t[(cb + 2) * LDSK + kr] = f2bf(v.z);
    t[(cb + 3) * LDSK + kr] = f2bf(v.w);
  }
  __syncthreads();

#pragma unroll
  for (int i = 0; i < 2; ++i) {
    int f  = tid + i * 256;        // 512 uint4 = 64 n-rows x 8 uint4
    int nr = f >> 3;
    int c  = f & 7;
    if (n0 + nr < N)
      *(uint4*)(Wt + (size_t)(n0 + nr) * K + k0 + c * 8) = *(const uint4*)(t + nr * LDSK + c * 8);
  }
}

// ---------------- bf16 GEMM: C = act(A @ Bt^T + bias) ----------------------
// A: bf16 [M][K]   Bt: bf16 [N][K]   bias: fp32 [N]

// issue one tile's async copies: exactly 8 instructions per wave (uniform
// count is REQUIRED for the s_wait_asynccnt thresholds; out-of-range B rows
// clamp the source address instead of branching so the instruction still
// issues -- the garbage only reaches output columns >= N which are not stored)
__device__ __forceinline__ void issue_tile_async(
    const unsigned short* __restrict__ A, const unsigned short* __restrict__ Bt,
    int m0, int n0, int N, int K, int k0, int tid,
    unsigned short* sA, unsigned short* sB) {
#pragma unroll
  for (int i = 0; i < 4; ++i) {
    int f = tid + i * 256;         // 1024 b128 = 128 rows x 8
    int row = f >> 3;
    int c   = f & 7;
    async_copy_b128(A + (size_t)(m0 + row) * K + k0 + c * 8,
                    sA + row * LDSK + c * 8);
    int col = n0 + row;
    int colc = (col < N) ? col : 0;                    // clamp, keep EXEC full
    async_copy_b128(Bt + (size_t)colc * K + k0 + c * 8,
                    sB + row * LDSK + c * 8);
  }
}

__device__ __forceinline__ void compute_tile(
    const unsigned short* sA, const unsigned short* sB, v8f acc[4][2],
    int waveM, int waveN, int l16, int lhalf) {
#pragma unroll
  for (int kc = 0; kc < 2; ++kc) {         // two K=32 chunks per TILE_K=64
    const int ko = kc * 32;
    FragU fa[4], fb[2];
#pragma unroll
    for (int mi = 0; mi < 4; ++mi) {
      // A operand (16x32, 16-bit): half 0 -> K {0..7,16..23}, half 1 -> {8..15,24..31}
      const unsigned short* p = sA + (waveM * 64 + mi * 16 + l16) * LDSK + ko + lhalf * 8;
      fa[mi].u[0] = *(const uint4*)(p);
      fa[mi].u[1] = *(const uint4*)(p + 16);
    }
#pragma unroll
    for (int ni = 0; ni < 2; ++ni) {
      // B operand (32x16, 16-bit): half 0 -> K {0..7,16..23}, half 1 -> {8..15,24..31}
      const unsigned short* p = sB + (waveN * 32 + ni * 16 + l16) * LDSK + ko + lhalf * 8;
      fb[ni].u[0] = *(const uint4*)(p);
      fb[ni].u[1] = *(const uint4*)(p + 16);
    }
#pragma unroll
    for (int mi = 0; mi < 4; ++mi)
#pragma unroll
      for (int ni = 0; ni < 2; ++ni)
        acc[mi][ni] = __builtin_amdgcn_wmma_f32_16x16x32_bf16(
            false, fa[mi].v, false, fb[ni].v, (short)0, acc[mi][ni], false, false);
  }
}

__global__ __launch_bounds__(256) void gemm_bf16_wmma(
    const unsigned short* __restrict__ A,   // bf16 [M][K]
    const unsigned short* __restrict__ Bt,  // bf16 [N][K]
    const float* __restrict__ bias,         // fp32 [N]
    float* __restrict__ Cf,                 // fp32 [M][N] or null
    unsigned short* __restrict__ Cb,        // bf16 [M][N] or null
    int M, int N, int K, int doRelu) {
  __shared__ unsigned short sA[3][TILE_M * LDSK];   // 3 x 18 KB
  __shared__ unsigned short sB[3][TILE_N * LDSK];   // 3 x 18 KB

  const int tid   = threadIdx.x;
  const int lane  = tid & 31;
  const int wid   = tid >> 5;
  const int waveN = wid & 3;
  const int waveM = wid >> 2;
  const int l16   = lane & 15;
  const int lhalf = lane >> 4;

  const int m0 = blockIdx.y * TILE_M;
  const int n0 = blockIdx.x * TILE_N;

  v8f acc[4][2];
#pragma unroll
  for (int mi = 0; mi < 4; ++mi)
#pragma unroll
    for (int ni = 0; ni < 2; ++ni)
      acc[mi][ni] = (v8f){0.f, 0.f, 0.f, 0.f, 0.f, 0.f, 0.f, 0.f};

  const int numT = K / TILE_K;

  // pipeline prologue: two tiles in flight (16 async instructions outstanding)
  issue_tile_async(A, Bt, m0, n0, N, K, 0, tid, sA[0], sB[0]);
  if (numT > 1)
    issue_tile_async(A, Bt, m0, n0, N, K, TILE_K, tid, sA[1], sB[1]);

  int bufC = 0;                  // buffer being computed (t % 3)
  for (int t = 0; t < numT; ++t) {
    // tile t landed when only the successor batch (8 instrs) is outstanding
    if (t < numT - 1) asm volatile("s_wait_asynccnt 8" ::: "memory");
    else              asm volatile("s_wait_asynccnt 0" ::: "memory");
    __syncthreads();             // all waves' portions of tile t are in LDS

    if (t + 2 < numT) {
      int bufN = bufC + 2; if (bufN >= 3) bufN -= 3;   // last read at iter t-1
      issue_tile_async(A, Bt, m0, n0, N, K, (t + 2) * TILE_K, tid, sA[bufN], sB[bufN]);
    }

    compute_tile(sA[bufC], sB[bufC], acc, waveM, waveN, l16, lhalf);
    if (++bufC == 3) bufC = 0;
  }

  // epilogue: VGPR r -> row = base + r + 8*lhalf, col = base + l16
#pragma unroll
  for (int mi = 0; mi < 4; ++mi) {
    int rowBase = m0 + waveM * 64 + mi * 16 + lhalf * 8;
#pragma unroll
    for (int ni = 0; ni < 2; ++ni) {
      int col = n0 + waveN * 32 + ni * 16 + l16;
      if (col < N) {
        float bv = bias[col];
#pragma unroll
        for (int r = 0; r < 8; ++r) {
          float v = acc[mi][ni][r] + bv;
          if (doRelu) v = fmaxf(v, 0.0f);
          if (Cf) Cf[(size_t)(rowBase + r) * N + col] = v;
          else    Cb[(size_t)(rowBase + r) * N + col] = f2bf(v);
        }
      }
    }
  }
}

// ---------------------------------------------------------------------------
extern "C" void kernel_launch(void* const* d_in, const int* in_sizes, int n_in,
                              void* d_out, int out_size, void* d_ws, size_t ws_size,
                              hipStream_t stream) {
  // inputs: features, batch_indices(unused: sorted -> identity), W1, b1, W2, b2, Wc, bc
  const float* X  = (const float*)d_in[0];
  const float* W1 = (const float*)d_in[2];
  const float* b1 = (const float*)d_in[3];
  const float* W2 = (const float*)d_in[4];
  const float* b2 = (const float*)d_in[5];
  const float* Wc = (const float*)d_in[6];
  const float* bc = (const float*)d_in[7];
  float* out = (float*)d_out;

  const int M = 2048, H = 1024, K1 = 7 * 7 * 256 /*12544*/, NC = 81;

  // workspace layout (bf16 elements)
  unsigned short* Xb  = (unsigned short*)d_ws;              // [M][K1]
  unsigned short* W1t = Xb  + (size_t)M * K1;               // [H][K1]
  unsigned short* W2t = W1t + (size_t)H * K1;               // [H][H]
  unsigned short* Wct = W2t + (size_t)H * H;                // [NC][H]
  unsigned short* H1  = Wct + (size_t)NC * H;               // [M][H]
  unsigned short* H2  = H1  + (size_t)M * H;                // [M][H]

  dim3 blk(256);

  // 1) convert X -> bf16
  {
    long n4 = ((long)M * K1) / 4;
    cvt_bf16_kernel<<<dim3((unsigned)((n4 + 255) / 256)), blk, 0, stream>>>(X, Xb, n4);
  }
  // 2) convert + transpose weights
  cvt_transpose_kernel<<<dim3(H / 64, K1 / 64), blk, 0, stream>>>(W1, W1t, K1, H);
  cvt_transpose_kernel<<<dim3(H / 64, H  / 64), blk, 0, stream>>>(W2, W2t, H, H);
  cvt_transpose_kernel<<<dim3((NC + 63) / 64, H / 64), blk, 0, stream>>>(Wc, Wct, H, NC);

  // 3) GEMM chain
  gemm_bf16_wmma<<<dim3(H / TILE_N, M / TILE_M), blk, 0, stream>>>(
      Xb, W1t, b1, nullptr, H1, M, H, K1, 1);
  gemm_bf16_wmma<<<dim3(H / TILE_N, M / TILE_M), blk, 0, stream>>>(
      H1, W2t, b2, nullptr, H2, M, H, H, 1);
  gemm_bf16_wmma<<<dim3(1, M / TILE_M), blk, 0, stream>>>(
      H2, Wct, bc, out, nullptr, M, NC, H, 0);
}
